// LossFunction_14688788152705
// MI455X (gfx1250) — matrix operand order repaired and checked by reading, not compiled, not run on hardware
//
#include <hip/hip_runtime.h>
#include <hip/hip_bf16.h>
#include <stdint.h>

#define B_N   16384
#define DIM   256
#define EPSF  1e-6f
#define DEPS2 (256.0f * 1e-6f * 1e-6f)
#define MINING_MARGIN 0.1f
#define MARGIN        0.3f

typedef __attribute__((ext_vector_type(16))) __bf16 v16bf;
typedef __attribute__((ext_vector_type(8)))  float  v8f;

// ---------------- workspace layout (bytes) ----------------
#define OFF_AHI  (size_t)0                      // 16384*256*2 = 8 MB
#define OFF_ALO  (size_t)(8u<<20)
#define OFF_PHI  (size_t)(16u<<20)
#define OFF_PLO  (size_t)(24u<<20)
#define OFF_NA2  (size_t)(32u<<20)              // 64 KB each
#define OFF_NP2  (OFF_NA2 + (64u<<10))
#define OFF_SA   (OFF_NA2 + 2*(64u<<10))
#define OFF_SP   (OFF_NA2 + 3*(64u<<10))
#define OFF_DG   (OFF_NA2 + 4*(64u<<10))
#define OFF_BEST (OFF_NA2 + 5*(64u<<10))        // 16384 * u64 = 128 KB
#define OFF_ACC  (OFF_BEST + (128u<<10))        // 1 float

__device__ __forceinline__ float wred32(float v) {
#pragma unroll
  for (int m = 16; m >= 1; m >>= 1) v += __shfl_xor(v, m, 32);
  return v;
}

__device__ __forceinline__ unsigned pcg(unsigned v) {
  unsigned s = v * 747796405u + 2891336453u;
  unsigned w = ((s >> ((s >> 28) + 4u)) ^ s) * 277803737u;
  return (w >> 22) ^ w;
}

// async DMA: 16 bytes global -> LDS, tracked by ASYNCcnt (GVS addressing mode)
__device__ __forceinline__ void async_b128(unsigned lds_off, const void* gbase,
                                           int byte_off) {
  asm volatile("global_load_async_to_lds_b128 %0, %1, %2"
               :: "v"(lds_off), "v"(byte_off), "s"(gbase)
               : "memory");
}

// ---------------- init: zero best[] and loss accumulator ----------------
__global__ void init_kernel(unsigned long long* __restrict__ best,
                            float* __restrict__ accum) {
  int idx = blockIdx.x * 256 + threadIdx.x;
  if (idx < B_N) best[idx] = 0ull;
  if (idx == 0) accum[0] = 0.0f;
}

// ---------------- prep: normalize + stats + bf16 hi/lo split ----------------
// one wave (32 lanes) per sample; lane handles 8 contiguous elements
__global__ void __launch_bounds__(256) prep_kernel(
    const float* __restrict__ x,
    __bf16* __restrict__ aHi, __bf16* __restrict__ aLo,
    __bf16* __restrict__ pHi, __bf16* __restrict__ pLo,
    float* __restrict__ na2, float* __restrict__ np2,
    float* __restrict__ sa,  float* __restrict__ sp,
    float* __restrict__ dg) {
  const int wave = threadIdx.x >> 5;
  const int lane = threadIdx.x & 31;
  const int i = blockIdx.x * 8 + wave;

  const float* xa = x + (size_t)i * (2 * DIM) + lane * 8;
  const float* xp = xa + DIM;
  float a[8], p[8];
  {
    float4 t0 = *(const float4*)xa, t1 = *(const float4*)(xa + 4);
    a[0]=t0.x; a[1]=t0.y; a[2]=t0.z; a[3]=t0.w; a[4]=t1.x; a[5]=t1.y; a[6]=t1.z; a[7]=t1.w;
    float4 u0 = *(const float4*)xp, u1 = *(const float4*)(xp + 4);
    p[0]=u0.x; p[1]=u0.y; p[2]=u0.z; p[3]=u0.w; p[4]=u1.x; p[5]=u1.y; p[6]=u1.z; p[7]=u1.w;
  }
  float ssA = 0.f, ssP = 0.f;
#pragma unroll
  for (int e = 0; e < 8; ++e) { ssA += a[e]*a[e]; ssP += p[e]*p[e]; }
  ssA = wred32(ssA); ssP = wred32(ssP);
  const float invA = 1.0f / fmaxf(sqrtf(ssA), 1e-12f);
  const float invP = 1.0f / fmaxf(sqrtf(ssP), 1e-12f);

  float vna = 0.f, vnp = 0.f, vsa = 0.f, vsp = 0.f, vdot = 0.f;
  union Pack { __bf16 h[8]; uint4 u; } hA, lA, hP, lP;
#pragma unroll
  for (int e = 0; e < 8; ++e) {
    float an = a[e] * invA, pn = p[e] * invP;
    vna += an*an; vnp += pn*pn; vsa += an; vsp += pn; vdot += an*pn;
    __bf16 ah = (__bf16)an; __bf16 ph = (__bf16)pn;
    hA.h[e] = ah; lA.h[e] = (__bf16)(an - (float)ah);
    hP.h[e] = ph; lP.h[e] = (__bf16)(pn - (float)ph);
  }
  const size_t off = (size_t)i * DIM + lane * 8;
  *(uint4*)(aHi + off) = hA.u;  *(uint4*)(aLo + off) = lA.u;
  *(uint4*)(pHi + off) = hP.u;  *(uint4*)(pLo + off) = lP.u;

  vna = wred32(vna); vnp = wred32(vnp);
  vsa = wred32(vsa); vsp = wred32(vsp); vdot = wred32(vdot);
  if (lane == 0) {
    na2[i] = vna; np2[i] = vnp; sa[i] = vsa; sp[i] = vsp;
    float D2 = vna + vnp - 2.0f * vdot + 2.0f * EPSF * (vsa - vsp) + DEPS2;
    dg[i] = sqrtf(fmaxf(D2, 0.0f));
  }
}

// ---------------- mine: fused WMMA GEMM + semihard mask + random argmax ----
// grid (128, 8): blockIdx.x = 128-row block, blockIdx.y = 2048-col group.
// 8 waves; wave w computes a 16x128 strip (8 16x16 WMMA subtiles).
// Async global->LDS DMA with double-buffered 128x32 panels.
__global__ void __launch_bounds__(256) mine_kernel(
    const __bf16* __restrict__ aHi, const __bf16* __restrict__ aLo,
    const __bf16* __restrict__ pHi, const __bf16* __restrict__ pLo,
    const float* __restrict__ na2, const float* __restrict__ np2,
    const float* __restrict__ sa,  const float* __restrict__ sp,
    const float* __restrict__ dg,
    unsigned long long* __restrict__ best) {
  // [buffer][AH, AL, BH, BL][512 x 16B chunks] = 64 KB
  __shared__ uint4 lds[2][4][512];

  const int tid  = threadIdx.x;
  const int wave = tid >> 5;
  const int lane = tid & 31;
  const int lm   = lane & 15;
  const int hh   = (lane >> 4) & 1;    // which K-half this lane owns in frags
  const int rowBase = blockIdx.x * 128;

  // per-row epilogue constants (row = rowBase + wave*16 + r + 8*hh)
  // D2 = ci[r] + cj - 2*dot ; mask on squared distances (sqrt-free)
  float ci[8], dg2[8], dgm2[8];
  int   iRow[8];
#pragma unroll
  for (int r = 0; r < 8; ++r) {
    int i = rowBase + wave * 16 + r + 8 * hh;
    iRow[r] = i;
    ci[r]   = na2[i] + 2.0f * EPSF * sa[i] + DEPS2;
    float d = dg[i];
    dg2[r]  = d * d;
    dgm2[r] = (d + MINING_MARGIN) * (d + MINING_MARGIN);
  }
  unsigned long long best64[8];
#pragma unroll
  for (int r = 0; r < 8; ++r) best64[r] = 0ull;

  // issue the 8 async DMAs (per wave) that stage one k-step's 4 panels
  auto prefetch = [&](int ks, int b, int colBase) {
#pragma unroll
    for (int c0 = 0; c0 < 2; ++c0) {
      int c = tid + c0 * 256;           // 512 16B chunks per panel
      int row = c >> 2, part = c & 3;
      int ga = (((rowBase + row) << 8) + ks * 32 + part * 8) * 2;  // bytes
      int gb = (((colBase + row) << 8) + ks * 32 + part * 8) * 2;
      async_b128((unsigned)(uintptr_t)&lds[b][0][c], aHi, ga);
      async_b128((unsigned)(uintptr_t)&lds[b][1][c], aLo, ga);
      async_b128((unsigned)(uintptr_t)&lds[b][2][c], pHi, gb);
      async_b128((unsigned)(uintptr_t)&lds[b][3][c], pLo, gb);
    }
  };

#pragma unroll 1
  for (int panel = 0; panel < 16; ++panel) {
    const int colBase = blockIdx.y * 2048 + panel * 128;
    v8f acc[8];
#pragma unroll
    for (int s = 0; s < 8; ++s)
#pragma unroll
      for (int e = 0; e < 8; ++e) acc[s][e] = 0.0f;

    prefetch(0, 0, colBase);

#pragma unroll 1
    for (int ks = 0; ks < 8; ++ks) {
      const int b = ks & 1;
      if (ks + 1 < 8) {
        prefetch(ks + 1, b ^ 1, colBase);          // overlap next stage's DMA
        asm volatile("s_wait_asynccnt 8" ::: "memory");  // current stage done
      } else {
        asm volatile("s_wait_asynccnt 0" ::: "memory");
      }
      __syncthreads();

      // A fragment: lane owns row (wave*16+lm); lanes<16 K 0..7/16..23, lanes>=16 K 8..15/24..31
      union U { uint4 u[2]; v16bf v; };
      const int arow = wave * 16 + lm;
      U ah, al;
      ah.u[0] = lds[b][0][arow * 4 + hh];  ah.u[1] = lds[b][0][arow * 4 + 2 + hh];
      al.u[0] = lds[b][1][arow * 4 + hh];  al.u[1] = lds[b][1][arow * 4 + 2 + hh];

#pragma unroll
      for (int s = 0; s < 8; ++s) {
        // B fragment: lane owns column (s*16+lm); lanes<16 K 0..7/16..23, lanes>=16 K 8..15/24..31
        const int bidx = (s * 16 + lm) * 4 + hh;
        U bh, bl;
        bh.u[0] = lds[b][2][bidx]; bh.u[1] = lds[b][2][bidx + 2];
        bl.u[0] = lds[b][3][bidx]; bl.u[1] = lds[b][3][bidx + 2];
        // split-precision fp32 emulation: hi*hi + hi*lo + lo*hi
        acc[s] = __builtin_amdgcn_wmma_f32_16x16x32_bf16(
            false, ah.v, false, bh.v, (short)0, acc[s], false, false);
        acc[s] = __builtin_amdgcn_wmma_f32_16x16x32_bf16(
            false, ah.v, false, bl.v, (short)0, acc[s], false, false);
        acc[s] = __builtin_amdgcn_wmma_f32_16x16x32_bf16(
            false, al.v, false, bh.v, (short)0, acc[s], false, false);
      }
      __syncthreads();
    }

    // fused epilogue: squared-distance mask + hashed random score (sqrt-free)
#pragma unroll
    for (int s = 0; s < 8; ++s) {
      const int j = colBase + s * 16 + lm;
      const float cj = np2[j] - 2.0f * EPSF * sp[j];
#pragma unroll
      for (int r = 0; r < 8; ++r) {
        float D2 = ci[r] + cj - 2.0f * acc[s][r];
        if (D2 > dg2[r] && D2 < dgm2[r]) {
          unsigned h = pcg((unsigned)iRow[r] * 16384u + (unsigned)j) | 1u;
          unsigned long long key = ((unsigned long long)h << 32) | (unsigned)j;
          if (key > best64[r]) best64[r] = key;
        }
      }
    }
  }

  // per-row max across the 16 lanes of each half, then one u64 atomic per row
#pragma unroll
  for (int r = 0; r < 8; ++r) {
    unsigned long long v = best64[r];
#pragma unroll
    for (int m = 8; m >= 1; m >>= 1) {
      unsigned long long o = __shfl_xor(v, m, 32);
      if (o > v) v = o;
    }
    if (lm == 0) atomicMax(&best[iRow[r]], v);   // lanes 0 and 16 -> rows r, r+8
  }
}

// ---------------- finalize: per-row triplet loss term ----------------
__global__ void __launch_bounds__(256) finalize_kernel(
    const float* __restrict__ x, const unsigned long long* __restrict__ best,
    const float* __restrict__ dg, float* __restrict__ accum) {
  const int wave = threadIdx.x >> 5;
  const int lane = threadIdx.x & 31;
  const int i = blockIdx.x * 8 + wave;

  unsigned long long b = best[i];
  unsigned j = (b == 0ull) ? (pcg((unsigned)i ^ 0x9E3779B9u) & (B_N - 1))
                           : (unsigned)(b & 0xFFFFFFFFull);

  const float* xa = x + (size_t)i * (2 * DIM) + lane * 8;
  const float* xn = x + (size_t)j * (2 * DIM) + DIM + lane * 8;
  float a[8], n[8];
  {
    float4 t0 = *(const float4*)xa, t1 = *(const float4*)(xa + 4);
    a[0]=t0.x; a[1]=t0.y; a[2]=t0.z; a[3]=t0.w; a[4]=t1.x; a[5]=t1.y; a[6]=t1.z; a[7]=t1.w;
    float4 u0 = *(const float4*)xn, u1 = *(const float4*)(xn + 4);
    n[0]=u0.x; n[1]=u0.y; n[2]=u0.z; n[3]=u0.w; n[4]=u1.x; n[5]=u1.y; n[6]=u1.z; n[7]=u1.w;
  }
  float ssA = 0.f, ssN = 0.f;
#pragma unroll
  for (int e = 0; e < 8; ++e) { ssA += a[e]*a[e]; ssN += n[e]*n[e]; }
  ssA = wred32(ssA); ssN = wred32(ssN);
  const float invA = 1.0f / fmaxf(sqrtf(ssA), 1e-12f);
  const float invN = 1.0f / fmaxf(sqrtf(ssN), 1e-12f);

  float nd2 = 0.f;
#pragma unroll
  for (int e = 0; e < 8; ++e) {
    float d = a[e] * invA - n[e] * invN + EPSF;
    nd2 += d * d;
  }
  nd2 = wred32(nd2);
  if (lane == 0) {
    float pd = dg[i];
    float term = fmaxf(pd * pd - nd2 + MARGIN, 0.0f);
    atomicAdd(accum, term);
  }
}

__global__ void scale_kernel(const float* __restrict__ accum, float* __restrict__ out) {
  out[0] = accum[0] * (1.0f / (float)B_N);
}

// ---------------- host launcher ----------------
extern "C" void kernel_launch(void* const* d_in, const int* in_sizes, int n_in,
                              void* d_out, int out_size, void* d_ws, size_t ws_size,
                              hipStream_t stream) {
  const float* x = (const float*)d_in[0];
  char* ws = (char*)d_ws;

  __bf16* aHi = (__bf16*)(ws + OFF_AHI);
  __bf16* aLo = (__bf16*)(ws + OFF_ALO);
  __bf16* pHi = (__bf16*)(ws + OFF_PHI);
  __bf16* pLo = (__bf16*)(ws + OFF_PLO);
  float*  na2 = (float*)(ws + OFF_NA2);
  float*  np2 = (float*)(ws + OFF_NP2);
  float*  sa  = (float*)(ws + OFF_SA);
  float*  sp  = (float*)(ws + OFF_SP);
  float*  dg  = (float*)(ws + OFF_DG);
  unsigned long long* best = (unsigned long long*)(ws + OFF_BEST);
  float* accum = (float*)(ws + OFF_ACC);
  float* out = (float*)d_out;

  init_kernel<<<B_N / 256, 256, 0, stream>>>(best, accum);
  prep_kernel<<<B_N / 8, 256, 0, stream>>>(x, aHi, aLo, pHi, pLo,
                                           na2, np2, sa, sp, dg);
  mine_kernel<<<dim3(B_N / 128, 8), 256, 0, stream>>>(aHi, aLo, pHi, pLo,
                                                      na2, np2, sa, sp, dg, best);
  finalize_kernel<<<B_N / 8, 256, 0, stream>>>(x, best, dg, accum);
  scale_kernel<<<1, 1, 0, stream>>>(accum, out);
}
